// SoftMultiFramePredictor_78572131713511
// MI455X (gfx1250) — matrix operand order repaired and checked by reading, not compiled, not run on hardware
//
#include <hip/hip_runtime.h>
#include <math.h>

typedef __attribute__((ext_vector_type(2)))  float    v2f;
typedef __attribute__((ext_vector_type(8)))  float    v8f;
typedef __attribute__((ext_vector_type(8)))  _Float16 v8h;
typedef __attribute__((ext_vector_type(16))) _Float16 v16h;
typedef __attribute__((ext_vector_type(4)))  unsigned int u32x4;
typedef __attribute__((ext_vector_type(4)))  int      i32x4;
typedef __attribute__((ext_vector_type(8)))  int      i32x8;

#define HH     128
#define WW     128
#define FEAT   128
#define PH     136      // padded rows  (4 halo each side)
#define PW     144      // padded cols  (4 left, 12 right for 32-wide WMMA tiles)
#define TEMPF  10.0f

// ---- workspace layout (bytes) ----
#define OFF_FA   ((size_t)0)                      // [128][128] f32  A_h  (stage2 A, row=i, col=m)
#define OFF_FB   ((size_t)65536)                  // [128][128] f32  B_h
#define OFF_FD   ((size_t)131072)                 // [128][128] f32  D_w^T (stage1 B, row=k, col=j)
#define OFF_FE   ((size_t)196608)                 // [128][128] f32  E_w^T
#define OFF_FILT ((size_t)262144)                 // [2][128][128][128] f32 filtered feats
#define OFF_TGTN (OFF_FILT + (size_t)16777216)    // [16384][128] f16 normalized tgt (pixel-major)
#define OFF_SRCN (OFF_TGTN + (size_t)4194304)     // [136][144][128] f16 normalized src, padded
#define OFF_SIMG (OFF_SRCN + (size_t)5013504)     // [3][136][136] f32 padded src image

#if defined(__has_builtin)
#if __has_builtin(__builtin_amdgcn_tensor_load_to_lds)
#define HAVE_TDM 1
#endif
#endif

static __device__ __forceinline__ v8f wmma_f32x4(v2f a, v2f b, v8f c) {
  return __builtin_amdgcn_wmma_f32_16x16x4_f32(false, a, false, b, (short)0, c, false, false);
}
static __device__ __forceinline__ v8f wmma_f16x32(v16h a, v16h b, v8f c) {
  return __builtin_amdgcn_wmma_f32_16x16x32_f16(false, a, false, b, (short)0, c, false, false);
}

#ifdef HAVE_TDM
// TDM: DMA a 2D f32 tile (tile_x cols x tile_y rows, row stride in elements) into LDS.
// D# layout per CDNA5 ISA (groups 2/3 zero => 2D tensor).
// clang-23 / therock form: 6 args (u32x4, i32x8, i32x4, i32x4, i32x8, i32 cpol).
static __device__ __forceinline__ void tdm_load_2d_f32(unsigned lds_off, const void* gptr,
                                                       unsigned tile_x, unsigned tile_y,
                                                       unsigned row_stride_elems) {
  unsigned long long ga = (unsigned long long)(uintptr_t)gptr;
  u32x4 g0 = { 1u,                                             // count=1, user desc
               lds_off,                                        // lds_addr (bytes)
               (unsigned)(ga & 0xffffffffu),                   // global_addr[31:0]
               ((unsigned)((ga >> 32) & 0x1ffffffu)) | 0x80000000u }; // addr[56:32] | type=2
  i32x8 g1;
  g1[0] = 0x00020000;                                 // workgroup_mask=0, data_size=2 (4B)
  g1[1] = (int)((tile_x & 0xffffu) << 16);            // tensor_dim0[15:0] = tile_x (no OOB)
  g1[2] = (int)((tile_y & 0xffffu) << 16);            // tensor_dim0[31:16]=0 | tensor_dim1[15:0]
  g1[3] = (int)((tile_x & 0xffffu) << 16);            // tensor_dim1[31:16]=0 | tile_dim0
  g1[4] = (int)(tile_y & 0xffffu);                    // tile_dim1 | tile_dim2=0
  g1[5] = (int)row_stride_elems;                      // tensor_dim0_stride[31:0]
  g1[6] = 0;
  g1[7] = 0;
  i32x4 z4 = { 0, 0, 0, 0 };
  i32x8 z8 = { 0, 0, 0, 0, 0, 0, 0, 0 };
  __builtin_amdgcn_tensor_load_to_lds(g0, g1, z4, z4, z8, 0);
}
#endif

// ---------------- K1: build the four circulant filter matrices ----------------
__global__ void k_filters(char* wsb) {
  int idx = blockIdx.x * 256 + threadIdx.x;            // 16384 entries
  if (idx >= 128 * 128) return;
  int r = idx >> 7, c = idx & 127;
  int d1 = (r - c) & 127;                              // for A_h/B_h  (row=i, col=m)
  int d2 = (c - r) & 127;                              // for D^T/E^T (row=k, col=j)
  const float w0 = 6.283185307179586f / 128.0f;
  float are = 0.f, aim = 0.f;
  for (int k = 0; k < 64; ++k) { float s, co; sincosf(w0 * (float)(k * d1), &s, &co); are += co; aim += s; }
  float dd = 1.0f, ee = 0.0f;
  for (int k = 1; k <= 32; ++k) { float s, co; sincosf(w0 * (float)(k * d2), &s, &co); dd += 2.0f * co; ee -= 2.0f * s; }
  const float sc = 1.0f / 128.0f;
  ((float*)(wsb + OFF_FA))[idx] = are * sc;
  ((float*)(wsb + OFF_FB))[idx] = aim * sc;
  ((float*)(wsb + OFF_FD))[idx] = dd * sc;
  ((float*)(wsb + OFF_FE))[idx] = ee * sc;
}

// ---------------- K2: spectral low-pass as two chained WMMA f32 GEMM stages ----------------
// per block: one (frame, channel) image. filtered = A_h*(X*D^T) + B_h*(X*E^T)
__global__ __launch_bounds__(256) void k_fftfilt(const float* __restrict__ feats, char* wsb) {
  __shared__ float T1s[128 * 16];
  __shared__ float T2s[128 * 16];
  __shared__ float BsD[128 * 16];   // FD[:, j0:j0+16] stripe, DMA'd by TDM
  __shared__ float BsE[128 * 16];   // FE[:, j0:j0+16]
  const int blk = blockIdx.x;                 // 0..255
  const int frame = blk >> 7, ch = blk & 127;
  const int tid = threadIdx.x, wave = tid >> 5, lane = tid & 31;
  const int hf = lane >> 4, tl = lane & 15;
  const float* X  = feats + (size_t)frame * 2097152 + (size_t)ch * 16384;
  const float* FA = (const float*)(wsb + OFF_FA);
  const float* FB = (const float*)(wsb + OFF_FB);
  const float* FD = (const float*)(wsb + OFF_FD);
  const float* FE = (const float*)(wsb + OFF_FE);
  float* FILT = (float*)(wsb + OFF_FILT) + (size_t)frame * 2097152 + (size_t)ch * 16384;

  // Hoist stage-1 A fragments (X rows of this wave's m-block) into registers:
  // they are reused unchanged by all 8 j-stripes.
  const int xrow = wave * 16 + tl;
  v2f xa[32];
  #pragma unroll
  for (int k = 0; k < 32; ++k) {
    const float* ap = X + (size_t)xrow * 128 + 4 * k + 2 * hf;
    v2f a; a.x = ap[0]; a.y = ap[1];
    xa[k] = a;
  }

  for (int jb = 0; jb < 8; ++jb) {
    const int j0 = jb * 16;
    // ---- stage the shared B stripes (identical for all waves) into LDS
#ifdef HAVE_TDM
    if (wave == 0) {
      tdm_load_2d_f32((unsigned)(uintptr_t)BsD, FD + j0, 16u, 128u, 128u);
      tdm_load_2d_f32((unsigned)(uintptr_t)BsE, FE + j0, 16u, 128u, 128u);
      __builtin_amdgcn_s_wait_tensorcnt(0);
    }
#else
    for (int q = tid; q < 2048; q += 256) {
      int rr = q >> 4, cc = q & 15;
      BsD[q] = FD[(size_t)rr * 128 + j0 + cc];
      BsE[q] = FE[(size_t)rr * 128 + j0 + cc];
    }
#endif
    __syncthreads();
    // ---- stage 1: wave w computes T1/T2 stripe rows [16w..16w+15], cols [j0..j0+15]
    for (int mat = 0; mat < 2; ++mat) {
      const float* Bs = mat ? BsE : BsD;
      float* Ts = mat ? T2s : T1s;
      v8f acc = {0.f, 0.f, 0.f, 0.f, 0.f, 0.f, 0.f, 0.f};
      #pragma unroll
      for (int k = 0; k < 32; ++k) {
        v2f b;
        b.x = Bs[(4 * k + 2 * hf) * 16 + tl];
        b.y = Bs[(4 * k + 2 * hf) * 16 + 16 + tl];
        acc = wmma_f32x4(xa[k], b, acc);
      }
      #pragma unroll
      for (int r = 0; r < 8; ++r) Ts[(wave * 16 + r + 8 * hf) * 16 + tl] = acc[r];
    }
    __syncthreads();
    // ---- stage 2: wave w computes output rows [16w..16w+15] of this j-stripe
    const int i0 = wave * 16;
    v8f acc = {0.f, 0.f, 0.f, 0.f, 0.f, 0.f, 0.f, 0.f};
    for (int m0 = 0; m0 < 128; m0 += 4) {
      v2f a; const float* ap = FA + (size_t)(i0 + tl) * 128 + m0 + 2 * hf;
      a.x = ap[0]; a.y = ap[1];
      v2f b; b.x = T1s[(m0 + 2 * hf) * 16 + tl]; b.y = T1s[(m0 + 2 * hf + 1) * 16 + tl];
      acc = wmma_f32x4(a, b, acc);
    }
    for (int m0 = 0; m0 < 128; m0 += 4) {
      v2f a; const float* ap = FB + (size_t)(i0 + tl) * 128 + m0 + 2 * hf;
      a.x = ap[0]; a.y = ap[1];
      v2f b; b.x = T2s[(m0 + 2 * hf) * 16 + tl]; b.y = T2s[(m0 + 2 * hf + 1) * 16 + tl];
      acc = wmma_f32x4(a, b, acc);
    }
    #pragma unroll
    for (int r = 0; r < 8; ++r) FILT[(size_t)(i0 + r + 8 * hf) * 128 + j0 + tl] = acc[r];
    __syncthreads();
  }
}

// ---------------- K3a: zero the padded buffers (halo must be exact 0) ----------------
__global__ void k_fill(char* wsb) {
  const size_t n = (5013504u + 221952u) / 4u;
  unsigned int* p = (unsigned int*)(wsb + OFF_SRCN);
  for (size_t idx = (size_t)blockIdx.x * 256 + threadIdx.x; idx < n; idx += (size_t)gridDim.x * 256) p[idx] = 0u;
}

// ---------------- K3b: copy src image into padded buffer ----------------
__global__ void k_img(const float* __restrict__ in_frames, char* wsb) {
  int idx = blockIdx.x * 256 + threadIdx.x;           // 3*128*128 = 49152
  if (idx >= 49152) return;
  int c = idx / 16384, rem = idx - c * 16384, ii = rem >> 7, jj = rem & 127;
  float* simg = (float*)(wsb + OFF_SIMG);
  simg[c * 18496 + (ii + 4) * 136 + (jj + 4)] = in_frames[idx];  // src = frame 0
}

// ---------------- K3c: channel L2-normalize, pack to f16 ----------------
__global__ __launch_bounds__(128) void k_norm(char* wsb) {
  const int frame = blockIdx.x >> 7, i = blockIdx.x & 127, j = threadIdx.x;
  const float* F = (const float*)(wsb + OFF_FILT) + (size_t)frame * 2097152 + (size_t)i * 128 + j;
  float s = 0.f;
  for (int c = 0; c < 128; ++c) { float v = F[(size_t)c * 16384]; s += v * v; }
  const float rinv = rsqrtf(s);
  if (frame == 1) {            // target
    _Float16* t = (_Float16*)(wsb + OFF_TGTN) + ((size_t)(i * 128 + j)) * 128;
    for (int c = 0; c < 128; ++c) t[c] = (_Float16)(F[(size_t)c * 16384] * rinv);
  } else {                     // source -> padded layout
    _Float16* sp = (_Float16*)(wsb + OFF_SRCN) + ((size_t)((i + 4) * PW + (j + 4))) * 128;
    for (int c = 0; c < 128; ++c) sp[c] = (_Float16)(F[(size_t)c * 16384] * rinv);
  }
}

// ---------------- K4: windowed correlation via f16 WMMA Gram bands + softmax + outputs ----------------
__global__ __launch_bounds__(256) void k_corr(char* wsb, float* __restrict__ out) {
  __shared__ float Wl[8][16][81];
  const int i = blockIdx.x;                           // target row
  const int tid = threadIdx.x, wave = tid >> 5, lane = tid & 31;
  const int hf = lane >> 4, tl = lane & 15;
  const int j0 = wave * 16;
  const _Float16* tgtn = (const _Float16*)(wsb + OFF_TGTN);
  const _Float16* srcn = (const _Float16*)(wsb + OFF_SRCN);
  const float*    simg = (const float*)(wsb + OFF_SIMG);

  // A fragments: 16 target pixels (row i, cols j0..j0+15), K = 128 channels in 4 steps of 32
  v16h afr[4];
  {
    const _Float16* tp = tgtn + ((size_t)(i * 128 + j0 + tl)) * 128;
    #pragma unroll
    for (int s = 0; s < 4; ++s) {
      v16h a = *(const v16h*)(tp + 32 * s + 16 * hf);
      afr[s] = a;
    }
  }

  for (int di = 0; di < 9; ++di) {
    const _Float16* srow = srcn + (size_t)(i + di) * PW * 128;   // padded row index
    if (di < 8) __builtin_prefetch(srcn + (size_t)(i + di + 1) * PW * 128 + (size_t)j0 * 128, 0, 1);
    v8f c0 = {0.f, 0.f, 0.f, 0.f, 0.f, 0.f, 0.f, 0.f};
    v8f c1 = {0.f, 0.f, 0.f, 0.f, 0.f, 0.f, 0.f, 0.f};
    #pragma unroll
    for (int s = 0; s < 4; ++s) {
      v16h b0 = *(const v16h*)(srow + ((size_t)(j0 + tl)) * 128 + 32 * s + 16 * hf);
      v16h b1 = *(const v16h*)(srow + ((size_t)(j0 + 16 + tl)) * 128 + 32 * s + 16 * hf);
      c0 = wmma_f16x32(afr[s], b0, c0);
      c1 = wmma_f16x32(afr[s], b1, c1);
    }
    // scatter the 9-wide diagonal band: weight(t, l) = G[t][t + l]
    #pragma unroll
    for (int r = 0; r < 8; ++r) {
      int t = r + 8 * hf;
      int l0 = tl - t;
      if (l0 >= 0 && l0 <= 8) Wl[wave][t][di * 9 + l0] = c0[r];
      int l1 = tl + 16 - t;
      if (l1 >= 0 && l1 <= 8) Wl[wave][t][di * 9 + l1] = c1[r];
    }
  }
  __syncthreads();

  if (lane < 16) {
    const int t = lane, j = j0 + t;
    float mx = -1e30f;
    for (int q = 0; q < 81; ++q) mx = fmaxf(mx, TEMPF * Wl[wave][t][q]);
    float ssum = 0.f;
    for (int q = 0; q < 81; ++q) { float e = __expf(TEMPF * Wl[wave][t][q] - mx); Wl[wave][t][q] = e; ssum += e; }
    const float inv = 1.0f / ssum;
    float p0 = 0.f, p1 = 0.f, en = 0.f, o0 = 0.f, o1 = 0.f, o2 = 0.f;
    float* wout = out + 81920 + (size_t)(i * 128 + j) * 81;
    for (int k = 0; k < 9; ++k) {
      for (int l = 0; l < 9; ++l) {
        float p = Wl[wave][t][k * 9 + l] * inv;
        wout[k * 9 + l] = p;
        float fl = (float)(l - 4), fk = (float)(k - 4);
        p0 += p * fl; p1 += p * fk;
        en += p * sqrtf(fl * fl + fk * fk);
        int rr = i + k, cc = j + l;                   // padded indices (halo 4)
        o0 += p * simg[0 * 18496 + rr * 136 + cc];
        o1 += p * simg[1 * 18496 + rr * 136 + cc];
        o2 += p * simg[2 * 18496 + rr * 136 + cc];
      }
    }
    const int pix = i * 128 + j;
    out[pix]             = o0;                        // out[0][0][0]
    out[16384 + pix]     = o1;
    out[32768 + pix]     = o2;
    out[49152 + pix * 2 + 0] = p0;                    // positions (flipped: l first)
    out[49152 + pix * 2 + 1] = p1;
    out[1409024 + pix]   = en;                        // exp_norm
  }
}

extern "C" void kernel_launch(void* const* d_in, const int* in_sizes, int n_in,
                              void* d_out, int out_size, void* d_ws, size_t ws_size,
                              hipStream_t stream) {
  const float* in_frames = (const float*)d_in[0];
  // d_in[1] (in_down) is unused by the reference computation.
  const float* feats = (const float*)d_in[2];
  float* out = (float*)d_out;
  char* wsb = (char*)d_ws;

  k_filters<<<64, 256, 0, stream>>>(wsb);
  k_fill  <<<1024, 256, 0, stream>>>(wsb);
  k_img   <<<192, 256, 0, stream>>>(in_frames, wsb);
  k_fftfilt<<<256, 256, 0, stream>>>(feats, wsb);
  k_norm  <<<256, 128, 0, stream>>>(wsb);
  k_corr  <<<128, 256, 0, stream>>>(wsb, out);
}
